// PlainNeRF_59072980189304
// MI455X (gfx1250) — matrix operand order repaired and checked by reading, not compiled, not run on hardware
//
#include <hip/hip_runtime.h>
#include <hip/hip_bf16.h>

typedef __attribute__((ext_vector_type(16))) _Float16 v16h;
typedef __attribute__((ext_vector_type(8)))  float    v8f;
typedef __attribute__((ext_vector_type(4)))  int      v4i;
typedef unsigned int u32;

#define STEPS 64
#define NRAYS (96 * 96)
#define WAVES 4
#define TPB (WAVES * 32)

// ---- layer tables ------------------------------------------------------
// layer:      0     1    2    3    4    5    6   7   8   9  10  11  12
//            Wi1  h1_0 h1_1 h1_2 h1_3  Wo1  Wi2 h2_0 h2_1 h2_2 h2_3 h2_4 Wo2
constexpr int LK[13]  = {3, 131, 128, 128, 131, 128, 34, 66, 32, 32, 66, 32, 32}; // source K (rows of W)
constexpr int LN[13]  = {128,128, 128, 128, 128, 33,  32, 32, 32, 32, 32, 32, 3}; // source N (cols of W)
constexpr int LNT[13] = {8, 8, 8, 8, 8, 3, 2, 2, 2, 2, 2, 2, 1};                  // N tiles of 16
constexpr int FOFF[13] = {0, 8, 48, 80, 112, 152, 164, 168, 174, 176, 178, 184, 186};
constexpr int BOFF[13] = {0, 128, 256, 384, 512, 640, 673, 705, 737, 769, 801, 833, 865};
constexpr int TOTAL_BIAS = 868;
constexpr int TOTAL_FRAGS = 187;          // sum(KT*NT); d_ws needs 187*1024 = 191,488 bytes

struct WPtrs  { const float* w[13]; };
struct Biases { const float* b[13]; };

// ---- async global -> LDS staging (CDNA5 path, guarded) -----------------
#if __has_builtin(__builtin_amdgcn_global_load_async_to_lds_b128) && \
    __has_builtin(__builtin_amdgcn_s_wait_asynccnt)
#define USE_ASYNC_LDS 1
#else
#define USE_ASYNC_LDS 0
#endif

typedef __attribute__((address_space(1))) v4i* gv4i_p;   // global int4*
typedef __attribute__((address_space(3))) v4i* lv4i_p;   // LDS int4*

__device__ __forceinline__ void copy16(const u32* g, u32* l) {
#if USE_ASYNC_LDS
    __builtin_amdgcn_global_load_async_to_lds_b128(
        (gv4i_p)(v4i*)(u32*)g, (lv4i_p)(v4i*)l, 0, 0);
#else
    // volatile: forbid LICM hoisting out of the tile loop (avoids mass spilling)
    const volatile uint4* s = (const volatile uint4*)g;
    uint4 v; v.x = s->x; v.y = s->y; v.z = s->z; v.w = s->w;
    *(uint4*)l = v;
#endif
}
__device__ __forceinline__ void wait_stage() {
#if USE_ASYNC_LDS
    __builtin_amdgcn_s_wait_asynccnt(0);
#endif
}

// ---- kernel 1: pre-swizzle f32 weights into wave32 WMMA B-fragments ----
// B-fragment (32x16 KxN) layout: lane<16: N=lane, K=0..15 (dword j: K=2j lo, 2j+1 hi);
// lane>=16: N=lane-16, K=16..31.  1 KB per fragment, lane-major 32B.
__global__ __launch_bounds__(32) void swizzle_kernel(WPtrs wp, u32* __restrict__ frags) {
    const int g    = blockIdx.x;
    const int lane = threadIdx.x;
    int layer = 0;
    #pragma unroll
    for (int l = 1; l < 13; ++l) if (g >= FOFF[l]) layer = l;
    const int local = g - FOFF[layer];
    const int nt = local % LNT[layer];
    const int kt = local / LNT[layer];
    const float* W = wp.w[layer];
    const int Ks = LK[layer], Ns = LN[layer];
    const int kbase = kt * 32 + ((lane >= 16) ? 16 : 0);
    const int col   = nt * 16 + (lane & 15);
    u32 d[8];
    #pragma unroll
    for (int j = 0; j < 8; ++j) {
        float v0 = 0.f, v1 = 0.f;
        const int r0 = kbase + 2 * j, r1 = r0 + 1;
        if (col < Ns) {
            if (r0 < Ks) v0 = W[r0 * Ns + col];
            if (r1 < Ks) v1 = W[r1 * Ns + col];
        }
        union { _Float16 h[2]; u32 u; } p;
        p.h[0] = (_Float16)v0; p.h[1] = (_Float16)v1;
        d[j] = p.u;
    }
    uint4* dst = (uint4*)(frags + (size_t)g * 256 + lane * 8);
    dst[0] = make_uint4(d[0], d[1], d[2], d[3]);
    dst[1] = make_uint4(d[4], d[5], d[6], d[7]);
}

// ---- fragment loaders --------------------------------------------------
// A-fragment (16x32 f16) from LDS: lane<16: row=lane, K{k0..k0+15};
// lane>=16: row=lane-16, K{k0+16..k0+31}
__device__ __forceinline__ v16h ldsA(const _Float16* rowp, int aCol) {
    union { uint4 u[2]; v16h v; } t;
    t.u[0] = *(const uint4*)(rowp + aCol);
    t.u[1] = *(const uint4*)(rowp + aCol + 8);
    return t.v;
}
__device__ __forceinline__ v16h ldB(const u32* p) {
    union { uint4 u[2]; v16h v; } t;
    const uint4* q = (const uint4*)p;
    t.u[0] = q[0]; t.u[1] = q[1];
    return t.v;
}

// MODE 0: hidden -> leaky relu, f16 to dst cols [0, NT*16)
// MODE 1: MLP1 out -> col0 = sigma to comp[..][3]; cols 1..32 -> act2 cols 34..65
// MODE 2: MLP2 out -> cols 0..2 = rgb to comp
// Block-cooperative: weights for k-chunk kt staged into LDS double buffer by
// all TPB threads (async), overlapped with WMMA on the previous chunk.
template <int KT, int NT, int MODE>
__device__ __forceinline__ void run_layer(
    const _Float16* src, int srcStride, int colStart,
    const u32* __restrict__ gfrag,        // global fragments, this layer
    const float* __restrict__ sbias, int validN,   // bias (in LDS)
    _Float16* dst, int dstStride,
    float* comp, int s0, int lane, int tid, u32* wstage)
{
    constexpr int CHUNKS = NT * 64;       // 16B chunks per k-slice (NT KB)
    const int sel   = (lane >= 16) ? 16 : 0;
    const int rowHi = (lane >= 16) ? 8 : 0;
    const int c0    = lane & 15;

    // stage k-chunk 0
    #pragma unroll
    for (int i = 0; i < (CHUNKS + TPB - 1) / TPB; ++i) {
        const int c = tid + i * TPB;
        if (c < CHUNKS) copy16(gfrag + c * 4, wstage + c * 4);
    }
    wait_stage();
    __syncthreads();

    v8f acc[NT];
    #pragma unroll
    for (int nt = 0; nt < NT; ++nt) {
        const int col = nt * 16 + c0;
        const float bv = (col < validN) ? sbias[col] : 0.f;
        acc[nt] = (v8f){bv, bv, bv, bv, bv, bv, bv, bv};
    }
    const _Float16* rowp = src + (size_t)c0 * srcStride + colStart;
    #pragma unroll
    for (int kt = 0; kt < KT; ++kt) {
        if (kt + 1 < KT) {                // prefetch next k-chunk into other buffer
            const u32* g = gfrag + (kt + 1) * NT * 256;
            u32* d = wstage + ((kt + 1) & 1) * 2048;
            #pragma unroll
            for (int i = 0; i < (CHUNKS + TPB - 1) / TPB; ++i) {
                const int c = tid + i * TPB;
                if (c < CHUNKS) copy16(g + c * 4, d + c * 4);
            }
        }
        const u32* wbuf = wstage + (kt & 1) * 2048;
        const v16h A = ldsA(rowp, kt * 32 + sel);
        #pragma unroll
        for (int nt = 0; nt < NT; ++nt) {
            const v16h B = ldB(wbuf + nt * 256 + lane * 8);
            acc[nt] = __builtin_amdgcn_wmma_f32_16x16x32_f16(
                false, A, false, B, (short)0, acc[nt], false, false);
        }
        wait_stage();
        __syncthreads();                  // staging done + all waves done reading wbuf
    }
    #pragma unroll
    for (int nt = 0; nt < NT; ++nt) {
        const int col = nt * 16 + c0;
        #pragma unroll
        for (int r = 0; r < 8; ++r) {
            float v = acc[nt][r];
            const int row = r + rowHi;
            if constexpr (MODE == 0) {
                v = (v >= 0.f) ? v : 0.01f * v;
                dst[(size_t)row * dstStride + col] = (_Float16)v;
            } else if constexpr (MODE == 1) {
                if (col == 0) comp[(s0 + row) * 4 + 3] = v;
                else if (col <= 32)
                    dst[(size_t)row * dstStride + 33 + col] = (_Float16)v;
            } else {
                if (col < 3) comp[(s0 + row) * 4 + col] = v;
            }
        }
    }
}

// ---- kernel 2: main NeRF forward. 1 wave = 1 ray (64 samples = 4 tiles of 16)
__global__ __launch_bounds__(TPB) void nerf_kernel(
    const float* __restrict__ rays, float* __restrict__ out,
    const u32* __restrict__ frags, Biases bs)
{
    // act1: [h(0..127) | x(128..130) | 0(131..159)], stride 160 halfs
    // act2: [h(0..31) | ea(32,33), inter(34..65) | 0(66..95)], stride 96 halfs
    __shared__ __align__(16) _Float16 sA1[WAVES][16 * 160];   // 20 KB
    __shared__ __align__(16) _Float16 sA2[WAVES][16 * 96];    // 12 KB
    __shared__ __align__(16) float    sC [WAVES][STEPS * 4];  //  4 KB
    __shared__ __align__(16) u32      sW [2 * 2048];          // 16 KB weight double buffer
    __shared__ __align__(16) float    sB [TOTAL_BIAS + 4];    // ~3.5 KB biases

    const int tid  = threadIdx.x;
    const int lane = tid & 31;
    const int wid  = tid >> 5;
    const int ray  = blockIdx.x * WAVES + wid;

    _Float16* a1  = sA1[wid];
    _Float16* a2  = sA2[wid];
    float*   comp = sC[wid];

    // stage all biases into LDS once
    for (int l = 0; l < 13; ++l)
        for (int i = tid; i < LN[l]; i += TPB) sB[BOFF[l] + i] = bs.b[l][i];

    { // zero activation buffers (pad columns stay zero forever)
        const uint4 z = make_uint4(0, 0, 0, 0);
        uint4* p1 = (uint4*)a1;
        #pragma unroll
        for (int i = 0; i < 10; ++i) p1[lane + i * 32] = z;
        uint4* p2 = (uint4*)a2;
        #pragma unroll
        for (int i = 0; i < 6; ++i) p2[lane + i * 32] = z;
    }

    const float ro0 = rays[ray * 6 + 0], ro1 = rays[ray * 6 + 1], ro2 = rays[ray * 6 + 2];
    const float rd0 = rays[ray * 6 + 3], rd1 = rays[ray * 6 + 4], rd2 = rays[ray * 6 + 5];
    const float lim = 1.0f - 1e-6f;
    const float c0v = fminf(fmaxf(rd0, -lim), lim);
    const float c1v = fminf(fmaxf(rd1, -lim), lim);
    const float c2v = fminf(fmaxf(rd2, -lim), lim);
    const float elev = asinf(c2v);
    const float azim = atan2f(c0v, c1v);
    if (lane < 16) {
        a2[lane * 96 + 32] = (_Float16)elev;
        a2[lane * 96 + 33] = (_Float16)azim;
    }
    const float dt = 4.0f / 63.0f;
    __syncthreads();   // biases staged

    #pragma unroll 1
    for (int t = 0; t < 4; ++t) {
        const int s0 = t * 16;
        if (lane < 16) {
            const float tt = 2.0f + (float)(s0 + lane) * dt;
            a1[lane * 160 + 128] = (_Float16)(ro0 + tt * rd0);
            a1[lane * 160 + 129] = (_Float16)(ro1 + tt * rd1);
            a1[lane * 160 + 130] = (_Float16)(ro2 + tt * rd2);
        }
        // ---- MLP1 ----
        run_layer<1, 8, 0>(a1, 160, 128, frags + FOFF[0] * 256, sB + BOFF[0], 128, a1, 160, comp, s0, lane, tid, sW);
        run_layer<5, 8, 0>(a1, 160, 0,   frags + FOFF[1] * 256, sB + BOFF[1], 128, a1, 160, comp, s0, lane, tid, sW);
        run_layer<4, 8, 0>(a1, 160, 0,   frags + FOFF[2] * 256, sB + BOFF[2], 128, a1, 160, comp, s0, lane, tid, sW);
        run_layer<4, 8, 0>(a1, 160, 0,   frags + FOFF[3] * 256, sB + BOFF[3], 128, a1, 160, comp, s0, lane, tid, sW);
        run_layer<5, 8, 0>(a1, 160, 0,   frags + FOFF[4] * 256, sB + BOFF[4], 128, a1, 160, comp, s0, lane, tid, sW);
        run_layer<4, 3, 1>(a1, 160, 0,   frags + FOFF[5] * 256, sB + BOFF[5], 33,  a2, 96,  comp, s0, lane, tid, sW);
        // ---- MLP2 ----
        run_layer<2, 2, 0>(a2, 96, 32,   frags + FOFF[6] * 256, sB + BOFF[6], 32,  a2, 96,  comp, s0, lane, tid, sW);
        run_layer<3, 2, 0>(a2, 96, 0,    frags + FOFF[7] * 256, sB + BOFF[7], 32,  a2, 96,  comp, s0, lane, tid, sW);
        run_layer<1, 2, 0>(a2, 96, 0,    frags + FOFF[8] * 256, sB + BOFF[8], 32,  a2, 96,  comp, s0, lane, tid, sW);
        run_layer<1, 2, 0>(a2, 96, 0,    frags + FOFF[9] * 256, sB + BOFF[9], 32,  a2, 96,  comp, s0, lane, tid, sW);
        run_layer<3, 2, 0>(a2, 96, 0,    frags + FOFF[10] * 256, sB + BOFF[10], 32, a2, 96, comp, s0, lane, tid, sW);
        run_layer<1, 2, 0>(a2, 96, 0,    frags + FOFF[11] * 256, sB + BOFF[11], 32, a2, 96, comp, s0, lane, tid, sW);
        run_layer<1, 1, 2>(a2, 96, 0,    frags + FOFF[12] * 256, sB + BOFF[12], 3,  a2, 96, comp, s0, lane, tid, sW);
    }

    // ---- volume compositing (per-wave scan over 64 steps, LDS broadcast) ----
    float T = 1.0f, A0 = 0.f, A1 = 0.f, A2 = 0.f;
    for (int s = 0; s < STEPS; ++s) {
        const float sig   = fmaxf(comp[s * 4 + 3], 0.0f);
        const float dist  = (s == STEPS - 1) ? 1e10f : dt;
        const float alpha = 1.0f - __expf(-sig * dist);
        const float w     = alpha * T;
        A0 += w * comp[s * 4 + 0];
        A1 += w * comp[s * 4 + 1];
        A2 += w * comp[s * 4 + 2];
        T *= fmaxf(1.0f - alpha, 1e-10f);
    }
    if (lane == 0) {
        out[ray * 3 + 0] = A0;
        out[ray * 3 + 1] = A1;
        out[ray * 3 + 2] = A2;
    }
}

extern "C" void kernel_launch(void* const* d_in, const int* in_sizes, int n_in,
                              void* d_out, int out_size, void* d_ws, size_t ws_size,
                              hipStream_t stream) {
    const float* rays = (const float*)d_in[0];
    WPtrs wp; Biases bs;
    for (int l = 0; l < 13; ++l) {
        wp.w[l] = (const float*)d_in[1 + 2 * l];
        bs.b[l] = (const float*)d_in[2 + 2 * l];
    }
    u32* frags = (u32*)d_ws;  // requires ws_size >= 187 KiB
    swizzle_kernel<<<dim3(TOTAL_FRAGS), dim3(32), 0, stream>>>(wp, frags);
    nerf_kernel<<<dim3(NRAYS / WAVES), dim3(TPB), 0, stream>>>(rays, (float*)d_out, frags, bs);
}
